// MultiHeadAttention_73151882985558
// MI455X (gfx1250) — matrix operand order repaired and checked by reading, not compiled, not run on hardware
//
#include <hip/hip_runtime.h>
#include <hip/hip_bf16.h>

// ---- problem constants (from reference) ----
#define D_MODEL 1024
#define N_HEAD  16
#define DH      64
#define SEQ     2048
#define BATCH   4
#define NTOK    (BATCH * SEQ)   // 8192

typedef __attribute__((ext_vector_type(16))) __bf16 v16bf;
typedef __attribute__((ext_vector_type(8)))  float  v8f;
typedef unsigned short u16;
typedef unsigned int   u32;
typedef int v4i __attribute__((vector_size(16)));

// gfx1250 async global->LDS path (ASYNCcnt), gated so compile never breaks
#if defined(__has_builtin)
#if __has_builtin(__builtin_amdgcn_global_load_async_to_lds_b128) && \
    __has_builtin(__builtin_amdgcn_s_wait_asynccnt)
#define HAVE_ASYNC 1
#endif
#endif

#ifdef HAVE_ASYNC
// builtin signature (from clang diagnostic): (v4i addrspace(1)*, v4i addrspace(3)*, imm offset, imm cpol)
#define ASYNC_B128(ldsp, glbp)                                              \
  __builtin_amdgcn_global_load_async_to_lds_b128(                           \
      (__attribute__((address_space(1))) v4i*)(glbp),                       \
      (__attribute__((address_space(3))) v4i*)(ldsp), 0, 0)
#define ASYNC_WAIT() __builtin_amdgcn_s_wait_asynccnt(0)
#endif

union FragBF { v16bf v; uint4 q[2]; u16 u[16]; };
union FragF  { v8f   v; float f[8]; };

__device__ __forceinline__ u16 f2bf(float f) {
  u32 u = __float_as_uint(f);
  u32 r = u + 0x7FFFu + ((u >> 16) & 1u);   // round-to-nearest-even
  return (u16)(r >> 16);
}

__device__ __forceinline__ float redmax16(float x) {
  #pragma unroll
  for (int m = 8; m >= 1; m >>= 1) x = fmaxf(x, __shfl_xor(x, m, 32));
  return x;  // reduced across each 16-lane half (rows live per-half)
}
__device__ __forceinline__ float redsum16(float x) {
  #pragma unroll
  for (int m = 8; m >= 1; m >>= 1) x += __shfl_xor(x, m, 32);
  return x;
}

// ---------------- fp32 -> bf16 convert (x) ----------------
__global__ void mha_cvt_bf16(const float* __restrict__ src, u16* __restrict__ dst, int n4) {
  int i = blockIdx.x * blockDim.x + threadIdx.x;
  if (i >= n4) return;
  float4 f = ((const float4*)src)[i];
  ushort4 o;
  o.x = f2bf(f.x); o.y = f2bf(f.y); o.z = f2bf(f.z); o.w = f2bf(f.w);
  ((ushort4*)dst)[i] = o;
}

// ------- fp32 W[k][n] -> bf16 Wt[n][k] (transpose once, reused 64x) -------
__global__ __launch_bounds__(256)
void mha_cvt_w_t(const float* __restrict__ W, u16* __restrict__ Wt) {
  __shared__ u16 lT[64][65];           // padded to dodge bank conflicts
  const int k0 = blockIdx.x * 64, n0 = blockIdx.y * 64;
  const int t = threadIdx.x;
  const int row = t >> 2, sg = t & 3;  // row 0..63, sg selects 16 cols
  #pragma unroll
  for (int i = 0; i < 16; i += 4) {
    float4 f = *(const float4*)(W + (size_t)(k0 + row) * D_MODEL + n0 + sg * 16 + i);
    lT[sg * 16 + i + 0][row] = f2bf(f.x);
    lT[sg * 16 + i + 1][row] = f2bf(f.y);
    lT[sg * 16 + i + 2][row] = f2bf(f.z);
    lT[sg * 16 + i + 3][row] = f2bf(f.w);
  }
  __syncthreads();
  u16 tmp[16];
  #pragma unroll
  for (int i = 0; i < 16; i++) tmp[i] = lT[row][sg * 16 + i];
  uint4* d = (uint4*)(Wt + (size_t)(n0 + row) * D_MODEL + k0 + sg * 16);
  d[0] = *(uint4*)&tmp[0];
  d[1] = *(uint4*)&tmp[8];
}

// ---------------- tiled bf16 WMMA GEMM ----------------
// C[8192,1024] = A[8192,1024] * Wt^T + bias   (Wt is [n][k], pre-transposed)
// mode 0,1 : store bf16 Q/K to head-split [B,H,S,dh]
// mode 2   : store bf16 V transposed to [B,H,dh,S] (flash stages it directly)
// mode 3   : store fp32 to outf (final output projection)
#define GM 128
#define GN 128
#define GK 64

__global__ __launch_bounds__(256)
void mha_gemm_bf16(const u16* __restrict__ A, const u16* __restrict__ Wt,
                   const float* __restrict__ bias, u16* __restrict__ outb,
                   float* __restrict__ outf, int mode)
{
#ifdef HAVE_ASYNC
  __shared__ u16 lA[2][GM][GK];   // double buffered: async fill overlaps compute
  __shared__ u16 lB[2][GN][GK];
#else
  __shared__ u16 lA[1][GM][GK];
  __shared__ u16 lB[1][GN][GK];
#endif

  const int tid  = threadIdx.x;
  const int lane = tid & 31;
  const int wave = tid >> 5;
  const int hi   = lane >> 4;
  const int lm   = lane & 15;
  const int m0   = blockIdx.x * GM;
  const int n0   = blockIdx.y * GN;
  const int wm   = (wave >> 2) * 64;   // 0 / 64
  const int wn   = (wave & 3) * 32;    // 0/32/64/96

  const int row = tid >> 1, half = tid & 1;   // staging: 2 threads/row, 32 u16 each
  const u16* srcA = A  + (size_t)(m0 + row) * D_MODEL + half * 32;
  const u16* srcB = Wt + (size_t)(n0 + row) * D_MODEL + half * 32;

  FragF acc[4][2];
  #pragma unroll
  for (int i = 0; i < 4; i++)
    #pragma unroll
    for (int j = 0; j < 2; j++)
      #pragma unroll
      for (int e = 0; e < 8; e++) acc[i][j].f[e] = 0.0f;

#ifdef HAVE_ASYNC
  // prologue: kick tile 0 into buffer 0
  #pragma unroll
  for (int i = 0; i < 4; i++) {
    ASYNC_B128(&lA[0][row][half * 32 + i * 8], srcA + i * 8);
    ASYNC_B128(&lB[0][row][half * 32 + i * 8], srcB + i * 8);
  }
#endif

  int it = 0;
  for (int kk = 0; kk < D_MODEL; kk += GK, ++it) {
    int bi;
#ifdef HAVE_ASYNC
    bi = it & 1;
    ASYNC_WAIT();          // my tile-t async writes landed in LDS
    __syncthreads();       // everyone's landed; prev buffer free for refill
    if (kk + GK < D_MODEL) {
      #pragma unroll
      for (int i = 0; i < 4; i++) {
        ASYNC_B128(&lA[bi ^ 1][row][half * 32 + i * 8], srcA + kk + GK + i * 8);
        ASYNC_B128(&lB[bi ^ 1][row][half * 32 + i * 8], srcB + kk + GK + i * 8);
      }
    }
#else
    bi = 0;
    __syncthreads();
    {
      const uint4* s = (const uint4*)(srcA + kk);
      uint4* d = (uint4*)&lA[0][row][half * 32];
      d[0] = s[0]; d[1] = s[1]; d[2] = s[2]; d[3] = s[3];
      const uint4* sw = (const uint4*)(srcB + kk);
      uint4* dw = (uint4*)&lB[0][row][half * 32];
      dw[0] = sw[0]; dw[1] = sw[1]; dw[2] = sw[2]; dw[3] = sw[3];
    }
    __syncthreads();
#endif

    #pragma unroll
    for (int kf = 0; kf < 2; kf++) {
      FragBF bf[2];
      #pragma unroll
      for (int nj = 0; nj < 2; nj++) {
        int n = wn + nj * 16 + lm;
        bf[nj].q[0] = *(const uint4*)&lB[bi][n][kf * 32 + hi * 8];
        bf[nj].q[1] = *(const uint4*)&lB[bi][n][kf * 32 + 16 + hi * 8];
      }
      #pragma unroll
      for (int mi = 0; mi < 4; mi++) {
        FragBF af;
        int m = wm + mi * 16 + lm;
        af.q[0] = *(const uint4*)&lA[bi][m][kf * 32 + hi * 8];
        af.q[1] = *(const uint4*)&lA[bi][m][kf * 32 + 16 + hi * 8];
        #pragma unroll
        for (int nj = 0; nj < 2; nj++) {
          acc[mi][nj].v = __builtin_amdgcn_wmma_f32_16x16x32_bf16(
              false, af.v, false, bf[nj].v, (short)0, acc[mi][nj].v, false, false);
        }
      }
    }
  }

  #pragma unroll
  for (int mi = 0; mi < 4; mi++)
    #pragma unroll
    for (int nj = 0; nj < 2; nj++) {
      int col = n0 + wn + nj * 16 + lm;
      float bv = bias[col];
      #pragma unroll
      for (int r = 0; r < 8; r++) {
        int mrow = m0 + wm + mi * 16 + r + 8 * hi;
        float val = acc[mi][nj].f[r] + bv;
        if (mode == 3) {
          outf[(size_t)mrow * D_MODEL + col] = val;
        } else {
          int b = mrow >> 11, s = mrow & (SEQ - 1);
          int h = col >> 6,  dc = col & (DH - 1);
          if (mode == 2)  // V transposed: [B,H,dh,S]
            outb[(((size_t)(b * N_HEAD + h)) * DH + dc) * SEQ + s] = f2bf(val);
          else            // Q,K: [B,H,S,dh]
            outb[(((size_t)(b * N_HEAD + h)) * SEQ + s) * DH + dc] = f2bf(val);
        }
      }
    }
}

// ---------------- flash attention ----------------
// Q/K: bf16 [B*H, S, DH]; V: bf16 [B*H, DH, S] (pre-transposed).
// Out: bf16 [B*S, D_MODEL] (col = h*64 + d).
#define QT 128
#define KT 64

__global__ __launch_bounds__(256)
void mha_flash_attn(const u16* __restrict__ Q, const u16* __restrict__ K,
                    const u16* __restrict__ V, u16* __restrict__ O)
{
#ifdef HAVE_ASYNC
  __shared__ u16 lK[2][KT][DH];    // double buffered
  __shared__ u16 lV[2][DH][KT];
#else
  __shared__ u16 lK[1][KT][DH];
  __shared__ u16 lV[1][DH][KT];
#endif
  __shared__ u16 lP[8][16][KT];    // per-wave P re-layout scratch

  const int tid  = threadIdx.x;
  const int lane = tid & 31;
  const int wave = tid >> 5;
  const int hi   = lane >> 4;
  const int lm   = lane & 15;
  const int bh   = blockIdx.y;                    // b*16 + h
  const int q0   = blockIdx.x * QT + wave * 16;

  const u16* qb = Q + (size_t)bh * SEQ * DH;
  const u16* kb = K + (size_t)bh * SEQ * DH;
  const u16* vt = V + (size_t)bh * DH * SEQ;      // [d][s]

  // resident Q fragments (two 16x32 fragments over dh=64)
  FragBF qf[2];
  #pragma unroll
  for (int kf = 0; kf < 2; kf++) {
    const u16* qr = qb + (size_t)(q0 + lm) * DH + kf * 32;
    qf[kf].q[0] = *(const uint4*)(qr + hi * 8);
    qf[kf].q[1] = *(const uint4*)(qr + 16 + hi * 8);
  }

  FragF o[4];
  float mrow[8], lrow[8];
  #pragma unroll
  for (int dj = 0; dj < 4; dj++)
    #pragma unroll
    for (int e = 0; e < 8; e++) o[dj].f[e] = 0.0f;
  #pragma unroll
  for (int r = 0; r < 8; r++) { mrow[r] = -INFINITY; lrow[r] = 0.0f; }

  const int srow = tid >> 2, ssg = tid & 3;       // staging coords
  const u16* kRow = kb + (size_t)srow * DH + ssg * 16;   // + kt*DH
  const u16* vRow = vt + (size_t)srow * SEQ + ssg * 16;  // + kt

#ifdef HAVE_ASYNC
  // prologue: tile 0 into buffer 0
  ASYNC_B128(&lK[0][srow][ssg * 16],     kRow);
  ASYNC_B128(&lK[0][srow][ssg * 16 + 8], kRow + 8);
  ASYNC_B128(&lV[0][srow][ssg * 16],     vRow);
  ASYNC_B128(&lV[0][srow][ssg * 16 + 8], vRow + 8);
#endif

  int it = 0;
  for (int kt = 0; kt < SEQ; kt += KT, ++it) {
    int bi;
#ifdef HAVE_ASYNC
    bi = it & 1;
    ASYNC_WAIT();
    __syncthreads();
    if (kt + KT < SEQ) {
      int nb = bi ^ 1;
      ASYNC_B128(&lK[nb][srow][ssg * 16],     kRow + (size_t)(kt + KT) * DH);
      ASYNC_B128(&lK[nb][srow][ssg * 16 + 8], kRow + (size_t)(kt + KT) * DH + 8);
      ASYNC_B128(&lV[nb][srow][ssg * 16],     vRow + kt + KT);
      ASYNC_B128(&lV[nb][srow][ssg * 16 + 8], vRow + kt + KT + 8);
    }
#else
    bi = 0;
    __syncthreads();
    {
      const uint4* sk = (const uint4*)(kRow + (size_t)kt * DH);
      *(uint4*)&lK[0][srow][ssg * 16]     = sk[0];
      *(uint4*)&lK[0][srow][ssg * 16 + 8] = sk[1];
      const uint4* sv = (const uint4*)(vRow + kt);
      *(uint4*)&lV[0][srow][ssg * 16]     = sv[0];
      *(uint4*)&lV[0][srow][ssg * 16 + 8] = sv[1];
    }
    __syncthreads();
#endif

    // scores: S = Q * K^T, 4 fragments of 16 keys
    FragF sc[4];
    #pragma unroll
    for (int nj = 0; nj < 4; nj++) {
      #pragma unroll
      for (int e = 0; e < 8; e++) sc[nj].f[e] = 0.0f;
      int key = nj * 16 + lm;
      #pragma unroll
      for (int kf = 0; kf < 2; kf++) {
        FragBF bf;
        bf.q[0] = *(const uint4*)&lK[bi][key][kf * 32 + hi * 8];
        bf.q[1] = *(const uint4*)&lK[bi][key][kf * 32 + 16 + hi * 8];
        sc[nj].v = __builtin_amdgcn_wmma_f32_16x16x32_bf16(
            false, qf[kf].v, false, bf.v, (short)0, sc[nj].v, false, false);
      }
    }

    // online softmax per query row (row = r + 8*hi within each 16-lane half)
    #pragma unroll
    for (int r = 0; r < 8; r++) {
      float mx = -INFINITY;
      #pragma unroll
      for (int nj = 0; nj < 4; nj++) {
        sc[nj].f[r] *= 0.125f;                 // 1/sqrt(dh)
        mx = fmaxf(mx, sc[nj].f[r]);
      }
      mx = redmax16(mx);
      float mn = fmaxf(mrow[r], mx);
      float alpha = __expf(mrow[r] - mn);
      float rs = 0.0f;
      #pragma unroll
      for (int nj = 0; nj < 4; nj++) {
        float p = __expf(sc[nj].f[r] - mn);
        rs += p;
        lP[wave][r + 8 * hi][nj * 16 + lm] = f2bf(p);
      }
      rs = redsum16(rs);
      lrow[r] = lrow[r] * alpha + rs;
      mrow[r] = mn;
      #pragma unroll
      for (int dj = 0; dj < 4; dj++) o[dj].f[r] *= alpha;
    }

    // P fragments (A layout) from per-wave LDS; DS ops are in-order per wave
    FragBF pf[2];
    #pragma unroll
    for (int kf = 0; kf < 2; kf++) {
      const u16* pr = &lP[wave][lm][kf * 32];
      pf[kf].q[0] = *(const uint4*)(pr + hi * 8);
      pf[kf].q[1] = *(const uint4*)(pr + 16 + hi * 8);
    }
    // O += P * V
    #pragma unroll
    for (int dj = 0; dj < 4; dj++) {
      int d = dj * 16 + lm;
      #pragma unroll
      for (int kf = 0; kf < 2; kf++) {
        FragBF bf;
        bf.q[0] = *(const uint4*)&lV[bi][d][kf * 32 + hi * 8];
        bf.q[1] = *(const uint4*)&lV[bi][d][kf * 32 + 16 + hi * 8];
        o[dj].v = __builtin_amdgcn_wmma_f32_16x16x32_bf16(
            false, pf[kf].v, false, bf.v, (short)0, o[dj].v, false, false);
      }
    }
  }

  // normalize and store to [B*S, D_MODEL] bf16
  const int b = bh >> 4, h = bh & 15;
  #pragma unroll
  for (int dj = 0; dj < 4; dj++) {
    int dc = dj * 16 + lm;
    #pragma unroll
    for (int r = 0; r < 8; r++) {
      int qrow = q0 + r + 8 * hi;
      size_t token = (size_t)b * SEQ + qrow;
      float val = o[dj].f[r] / lrow[r];
      O[token * D_MODEL + h * DH + dc] = f2bf(val);
    }
  }
}

// ---------------- launch ----------------
extern "C" void kernel_launch(void* const* d_in, const int* in_sizes, int n_in,
                              void* d_out, int out_size, void* d_ws, size_t ws_size,
                              hipStream_t stream) {
  const float* x  = (const float*)d_in[0];
  const float* wq = (const float*)d_in[1];
  const float* bq = (const float*)d_in[2];
  const float* wk = (const float*)d_in[3];
  const float* bk = (const float*)d_in[4];
  const float* wv = (const float*)d_in[5];
  const float* bv = (const float*)d_in[6];
  const float* wo = (const float*)d_in[7];
  const float* bo = (const float*)d_in[8];
  float* out = (float*)d_out;

  // workspace layout (bf16 scratch, 88 MiB total)
  char* ws = (char*)d_ws;
  u16* xb  = (u16*)(ws);
  u16* wqb = (u16*)(ws + (size_t)16 * (1 << 20));
  u16* wkb = (u16*)(ws + (size_t)18 * (1 << 20));
  u16* wvb = (u16*)(ws + (size_t)20 * (1 << 20));
  u16* wob = (u16*)(ws + (size_t)22 * (1 << 20));
  u16* qbf = (u16*)(ws + (size_t)24 * (1 << 20));
  u16* kbf = (u16*)(ws + (size_t)40 * (1 << 20));
  u16* vbf = (u16*)(ws + (size_t)56 * (1 << 20));
  u16* att = (u16*)(ws + (size_t)72 * (1 << 20));

  const int nX = NTOK * D_MODEL;      // 8,388,608
  mha_cvt_bf16<<<nX / 4 / 256, 256, 0, stream>>>(x, xb, nX / 4);

  dim3 wt(D_MODEL / 64, D_MODEL / 64);  // 16 x 16
  mha_cvt_w_t<<<wt, 256, 0, stream>>>(wq, wqb);
  mha_cvt_w_t<<<wt, 256, 0, stream>>>(wk, wkb);
  mha_cvt_w_t<<<wt, 256, 0, stream>>>(wv, wvb);
  mha_cvt_w_t<<<wt, 256, 0, stream>>>(wo, wob);

  dim3 gg(NTOK / GM, D_MODEL / GN);   // 64 x 8
  mha_gemm_bf16<<<gg, 256, 0, stream>>>(xb, wqb, bq, qbf, nullptr, 0);
  mha_gemm_bf16<<<gg, 256, 0, stream>>>(xb, wkb, bk, kbf, nullptr, 1);
  mha_gemm_bf16<<<gg, 256, 0, stream>>>(xb, wvb, bv, vbf, nullptr, 2);

  dim3 fg(SEQ / QT, BATCH * N_HEAD);  // 16 x 64
  mha_flash_attn<<<fg, 256, 0, stream>>>(qbf, kbf, vbf, att);

  mha_gemm_bf16<<<gg, 256, 0, stream>>>(att, wob, bo, nullptr, out, 3);
}
